// SelfMultiHeadAttention_74947179315615
// MI455X (gfx1250) — matrix operand order repaired and checked by reading, not compiled, not run on hardware
//
#include <hip/hip_runtime.h>
#include <hip/hip_bf16.h>
#include <stdint.h>

typedef __bf16 bf16t;
typedef __attribute__((ext_vector_type(16))) __bf16 v16bf;
typedef __attribute__((ext_vector_type(8)))  float  v8f;

// ---------------------------------------------------------------------------
// WMMA helper: D = A(16x32 bf16) * B(32x16 bf16) + C(16x16 f32)
// ---------------------------------------------------------------------------
__device__ __forceinline__ v8f wmma_bf16(v16bf a, v16bf b, v8f c) {
    return __builtin_amdgcn_wmma_f32_16x16x32_bf16(
        /*neg_a=*/false, a, /*neg_b=*/false, b,
        /*c_mod=*/(short)0, c, /*reuse_a=*/false, /*reuse_b=*/false);
}

// A fragment (16x32, 16-bit): lane L holds row M=L%16.
// lanes 0-15: elems 0..7 = K0..7,  elems 8..15 = K16..23
// lanes16-31: elems 0..7 = K8..15, elems 8..15 = K24..31
__device__ __forceinline__ v16bf load_a_frag(const bf16t* base, int ld, int lane) {
    const bf16t* p = base + (size_t)(lane & 15) * ld + ((lane >> 4) << 3);
    v16bf f;
    ((uint4*)&f)[0] = *(const uint4*)(p);
    ((uint4*)&f)[1] = *(const uint4*)(p + 16);
    return f;
}

// B fragment (32x16, 16-bit): lane L holds column N=L%16.
// lanes 0-15: elems 0..7 = K0..7, elems 8..15 = K16..23 ; lanes 16-31: elems 0..7 = K8..15, elems 8..15 = K24..31.
// Source is "N-major" (row n contiguous in K): two 16B reads per lane (K offsets 0 and 16).
__device__ __forceinline__ v16bf load_b_frag(const bf16t* base, int ld, int lane) {
    const bf16t* p = base + (size_t)(lane & 15) * ld + ((lane >> 4) << 3);
    v16bf f;
    ((uint4*)&f)[0] = *(const uint4*)(p);
    ((uint4*)&f)[1] = *(const uint4*)(p + 16);
    return f;
}

// ds_swizzle xor within 16-lane halves (C-layout row reductions)
template <int MASK>
__device__ __forceinline__ float swz_xor(float v) {
    int x = __builtin_amdgcn_ds_swizzle(__float_as_int(v), (MASK << 10) | 0x1f);
    return __int_as_float(x);
}

// Async global -> LDS copy, 16B per lane (ASYNCcnt-tracked, CDNA5 TDM-lite path).
// LDS address = low 32 bits of the flat shared-pointer (aperture truncation).
__device__ __forceinline__ void async_b128(void* ldsPtr, const void* gPtr) {
    uint32_t lds = (uint32_t)(uintptr_t)ldsPtr;
    asm volatile("global_load_async_to_lds_b128 %0, %1, off"
                 :: "v"(lds), "v"(gPtr) : "memory");
}
__device__ __forceinline__ void wait_async0() {
    asm volatile("s_wait_asynccnt 0x0" ::: "memory");
}

// ---------------------------------------------------------------------------
// f32 -> bf16 conversion
// ---------------------------------------------------------------------------
__global__ __launch_bounds__(256) void cvt_bf16_kernel(const float* __restrict__ in,
                                                       bf16t* __restrict__ out,
                                                       long n) {
    long i = (long)blockIdx.x * blockDim.x + threadIdx.x;
    long stride = (long)gridDim.x * blockDim.x;
    for (; i < n; i += stride) out[i] = (bf16t)in[i];
}

// ---------------------------------------------------------------------------
// f32 [K][N] -> bf16 transposed [N][K] (LDS-tiled 32x32)
// ---------------------------------------------------------------------------
__global__ __launch_bounds__(256) void cvt_transpose_kernel(const float* __restrict__ in,
                                                            bf16t* __restrict__ out,
                                                            int K, int N) {
    __shared__ float tile[32][33];
    const int kb = blockIdx.y * 32, nb = blockIdx.x * 32;
    const int tx = threadIdx.x & 31, ty = threadIdx.x >> 5;   // 32 x 8
#pragma unroll
    for (int i = 0; i < 32; i += 8)
        tile[ty + i][tx] = in[(size_t)(kb + ty + i) * N + nb + tx];
    __syncthreads();
#pragma unroll
    for (int i = 0; i < 32; i += 8)
        out[(size_t)(nb + ty + i) * K + kb + tx] = (bf16t)tile[tx][ty + i];
}

// ---------------------------------------------------------------------------
// WMMA GEMM (TN): C(f32)[M,N] = A(bf16)[M,K] * Bt(bf16)[N,K]^T
// Block tile 128x128, 8 waves (2x4), wave tile 64x32 (4x2 WMMA tiles).
// Double-buffered LDS filled with global_load_async_to_lds_b128.
// ---------------------------------------------------------------------------
__global__ __launch_bounds__(256) void gemm_bf16_tn(const bf16t* __restrict__ A,
                                                    const bf16t* __restrict__ Bt,
                                                    float* __restrict__ C,
                                                    int M, int N, int Kd) {
    (void)M;
    __shared__ __align__(16) bf16t sA[2][128][32];
    __shared__ __align__(16) bf16t sB[2][128][32];

    const int t    = threadIdx.x;
    const int lane = t & 31;
    const int wave = t >> 5;
    const int wm   = wave & 1;        // 2 waves along M (64 rows each)
    const int wn   = wave >> 1;       // 4 waves along N (32 cols each)
    const int mBase = blockIdx.y * 128;
    const int nBase = blockIdx.x * 128;

    // staging assignment: thread -> (row 0..127, 16-elem K half)
    const int row = t >> 1, kh = (t & 1) << 4;
    const bf16t* gA = A  + (size_t)(mBase + row) * Kd + kh;
    const bf16t* gB = Bt + (size_t)(nBase + row) * Kd + kh;

    v8f vz = {0.f, 0.f, 0.f, 0.f, 0.f, 0.f, 0.f, 0.f};
    v8f acc[4][2];
#pragma unroll
    for (int mt = 0; mt < 4; ++mt) { acc[mt][0] = vz; acc[mt][1] = vz; }

    auto fill = [&](int buf, int kb) {
        async_b128(&sA[buf][row][kh],     gA + kb);
        async_b128(&sA[buf][row][kh + 8], gA + kb + 8);
        async_b128(&sB[buf][row][kh],     gB + kb);
        async_b128(&sB[buf][row][kh + 8], gB + kb + 8);
    };

    const int nIter = Kd >> 5;        // BK = 32
    fill(0, 0);
    for (int it = 0; it < nIter; ++it) {
        wait_async0();                // my current-buffer fill landed in LDS
        __syncthreads();              // everyone's fill landed; prev reads done
        if (it + 1 < nIter) fill((it + 1) & 1, (it + 1) << 5);

        const bf16t* a0 = &sA[it & 1][wm * 64][0];
        const bf16t* b0 = &sB[it & 1][wn * 32][0];
        v16bf af[4], bfr[2];
#pragma unroll
        for (int mt = 0; mt < 4; ++mt) af[mt] = load_a_frag(a0 + mt * 16 * 32, 32, lane);
#pragma unroll
        for (int nt = 0; nt < 2; ++nt) bfr[nt] = load_b_frag(b0 + nt * 16 * 32, 32, lane);
#pragma unroll
        for (int mt = 0; mt < 4; ++mt)
#pragma unroll
            for (int nt = 0; nt < 2; ++nt)
                acc[mt][nt] = wmma_bf16(af[mt], bfr[nt], acc[mt][nt]);
    }

    // C layout: VGPR r -> row m=r (lanes 0-15) / m=r+8 (lanes 16-31), col n=lane%16
    const int cn = nBase + wn * 32 + (lane & 15);
    const int cm = mBase + wm * 64 + ((lane >> 4) << 3);
#pragma unroll
    for (int mt = 0; mt < 4; ++mt)
#pragma unroll
        for (int nt = 0; nt < 2; ++nt)
#pragma unroll
            for (int r = 0; r < 8; ++r)
                C[(size_t)(cm + mt * 16 + r) * N + cn + nt * 16] = acc[mt][nt][r];
}

// ---------------------------------------------------------------------------
// RoPE + split qkv(f32)[B*S,3072] -> Q[b,h,s,hd] K[b,kvh,s,hd] VT[b,kvh,hd,s] bf16
// ---------------------------------------------------------------------------
__global__ __launch_bounds__(256) void rope_split_kernel(const float* __restrict__ qkv,
                                                         bf16t* __restrict__ Q,
                                                         bf16t* __restrict__ K,
                                                         bf16t* __restrict__ VT) {
    const int bs = blockIdx.x;
    const int b = bs >> 11, s = bs & 2047;
    const float* row = qkv + (size_t)bs * 3072;
    const float LN1E4 = 9.210340371976184f;   // ln(10000)

    for (int item = threadIdx.x; item < 1792; item += 256) {
        if (item < 1280) {               // 1280 rotated pairs (q: 1024, k: 256)
            int col = item << 1;
            int hd  = col & 63;
            int i   = hd >> 1;
            float ang = (float)s * __expf(-(float)i * (LN1E4 / 32.0f));
            float sn, c;
            __sincosf(ang, &sn, &c);
            float x0 = row[col], x1 = row[col + 1];
            float y0 = x0 * c - x1 * sn;
            float y1 = x1 * c + x0 * sn;
            if (col < 2048) {
                int hh = col >> 6;
                bf16t* dst = Q + (((size_t)(b * 32 + hh) * 2048) + s) * 64 + hd;
                dst[0] = (bf16t)y0; dst[1] = (bf16t)y1;
            } else {
                int kc = col - 2048;
                int kvh = kc >> 6;
                bf16t* dst = K + (((size_t)(b * 8 + kvh) * 2048) + s) * 64 + (kc & 63);
                dst[0] = (bf16t)y0; dst[1] = (bf16t)y1;
            }
        } else {                          // 512 v elements, stored transposed
            int e = item - 1280;
            int kvh = e >> 6, hd = e & 63;
            VT[(((size_t)(b * 8 + kvh) * 64) + hd) * 2048 + s] = (bf16t)row[2560 + e];
        }
    }
}

// ---------------------------------------------------------------------------
// Flash attention (causal, GQA): one wave per (b, head, 16-row Q tile).
// ---------------------------------------------------------------------------
__global__ __launch_bounds__(256) void attn_kernel(const bf16t* __restrict__ Q,
                                                   const bf16t* __restrict__ Km,
                                                   const bf16t* __restrict__ VT,
                                                   bf16t* __restrict__ O) {
    __shared__ __align__(16) bf16t plds[8][16][32];   // per-wave P transpose buffer

    const int lane  = threadIdx.x & 31;
    const int wslot = threadIdx.x >> 5;
    const int wid   = blockIdx.x * 8 + wslot;         // 8192 waves total
    const int qt  = wid & 127;                        // q tile (16 rows each)
    const int h   = (wid >> 7) & 31;                  // q head
    const int b   = wid >> 12;                        // batch
    const int kvh = h >> 2;                           // kv head (group of 4)

    const bf16t* Qp = Q  + (((size_t)(b * 32 + h)   * 2048) + qt * 16) * 64;
    const bf16t* Kp = Km + (size_t)(b * 8 + kvh) * 2048 * 64;   // [t][hd]
    const bf16t* Vp = VT + (size_t)(b * 8 + kvh) * 64 * 2048;   // [hd][t]

    v16bf qf0 = load_a_frag(Qp,      64, lane);       // hd 0..31
    v16bf qf1 = load_a_frag(Qp + 32, 64, lane);       // hd 32..63

    v8f vz = {0.f, 0.f, 0.f, 0.f, 0.f, 0.f, 0.f, 0.f};
    v8f oacc[4] = {vz, vz, vz, vz};
    float mrow[8], rsum[8];
#pragma unroll
    for (int r = 0; r < 8; ++r) { mrow[r] = -1e30f; rsum[r] = 0.f; }

    bf16t* pl = &plds[wslot][0][0];
    const int hi8  = (lane >> 4) << 3;
    const int colt = lane & 15;
    const int nblocks = (qt >> 1) + 1;   // 32-key blocks up to the causal diagonal

    for (int kb2 = 0; kb2 < nblocks; ++kb2) {
        const int tBase = kb2 * 32;

        if (kb2 + 1 < nblocks) {         // prefetch next K/V block
            __builtin_prefetch(Kp + (size_t)(tBase + 32) * 64, 0, 1);
            __builtin_prefetch(Vp + tBase + 32, 0, 1);
        }

        // scores: two 16-key tiles, K-dim = 64 (two WMMA each)
        v8f s0 = vz, s1 = vz;
        s0 = wmma_bf16(qf0, load_b_frag(Kp + (size_t)tBase * 64,             64, lane), s0);
        s0 = wmma_bf16(qf1, load_b_frag(Kp + (size_t)tBase * 64 + 32,        64, lane), s0);
        s1 = wmma_bf16(qf0, load_b_frag(Kp + (size_t)(tBase + 16) * 64,      64, lane), s1);
        s1 = wmma_bf16(qf1, load_b_frag(Kp + (size_t)(tBase + 16) * 64 + 32, 64, lane), s1);

        float corr[8];
#pragma unroll
        for (int r = 0; r < 8; ++r) {
            int qrow = qt * 16 + hi8 + r;
            float a0 = (tBase + colt      <= qrow) ? s0[r] * 0.125f : -1e30f;
            float a1 = (tBase + 16 + colt <= qrow) ? s1[r] * 0.125f : -1e30f;
            float mx = fmaxf(a0, a1);
            mx = fmaxf(mx, swz_xor<1>(mx));
            mx = fmaxf(mx, swz_xor<2>(mx));
            mx = fmaxf(mx, swz_xor<4>(mx));
            mx = fmaxf(mx, swz_xor<8>(mx));
            float nm = fmaxf(mrow[r], mx);
            corr[r]  = __expf(mrow[r] - nm);
            mrow[r]  = nm;
            float p0 = __expf(a0 - nm);
            float p1 = __expf(a1 - nm);
            rsum[r]  = rsum[r] * corr[r] + p0 + p1;
            // transpose P (C layout -> row-major 16x32) through LDS
            pl[(hi8 + r) * 32 + colt]      = (bf16t)p0;
            pl[(hi8 + r) * 32 + 16 + colt] = (bf16t)p1;
        }
#pragma unroll
        for (int j = 0; j < 4; ++j)
#pragma unroll
            for (int r = 0; r < 8; ++r) oacc[j][r] *= corr[r];

        v16bf pf = load_a_frag(pl, 32, lane);          // P as A fragment (16x32)
#pragma unroll
        for (int j = 0; j < 4; ++j)
            oacc[j] = wmma_bf16(pf,
                load_b_frag(Vp + (size_t)(j * 16) * 2048 + tBase, 2048, lane), oacc[j]);
    }

    float inv[8];
#pragma unroll
    for (int r = 0; r < 8; ++r) {
        float sum = rsum[r];
        sum += swz_xor<1>(sum);
        sum += swz_xor<2>(sum);
        sum += swz_xor<4>(sum);
        sum += swz_xor<8>(sum);
        inv[r] = 1.0f / sum;
    }

    bf16t* Op = O + ((size_t)b * 2048 + qt * 16) * 2048 + h * 64;
#pragma unroll
    for (int j = 0; j < 4; ++j)
#pragma unroll
        for (int r = 0; r < 8; ++r)
            Op[(size_t)(hi8 + r) * 2048 + j * 16 + colt] = (bf16t)(oacc[j][r] * inv[r]);
}

// ---------------------------------------------------------------------------
// launch
// ---------------------------------------------------------------------------
extern "C" void kernel_launch(void* const* d_in, const int* in_sizes, int n_in,
                              void* d_out, int out_size, void* d_ws, size_t ws_size,
                              hipStream_t stream) {
    (void)in_sizes; (void)n_in; (void)out_size; (void)ws_size;
    const float* x    = (const float*)d_in[0];   // [2,2048,2048]
    const float* wqkv = (const float*)d_in[1];   // [2048,3072]
    const float* wout = (const float*)d_in[2];   // [2048,2048]
    float* out = (float*)d_out;                  // [2,2048,2048]

    char* ws = (char*)d_ws;
    size_t off = 0;
    bf16t* xbf   = (bf16t*)(ws + off); off += (size_t)4096 * 2048 * 2;
    bf16t* wqkvT = (bf16t*)(ws + off); off += (size_t)3072 * 2048 * 2;   // [N][K]
    bf16t* woutT = (bf16t*)(ws + off); off += (size_t)2048 * 2048 * 2;   // [N][K]
    float* qkvf  = (float*)(ws + off); off += (size_t)4096 * 3072 * 4;
    bf16t* Qb    = (bf16t*)(ws + off); off += (size_t)2 * 32 * 2048 * 64 * 2;
    bf16t* Kb    = (bf16t*)(ws + off); off += (size_t)2 * 8 * 2048 * 64 * 2;
    bf16t* VTb   = (bf16t*)(ws + off); off += (size_t)2 * 8 * 64 * 2048 * 2;
    bf16t* attnO = (bf16t*)(ws + off); off += (size_t)4096 * 2048 * 2;

    cvt_bf16_kernel<<<2048, 256, 0, stream>>>(x, xbf, 4096L * 2048);
    cvt_transpose_kernel<<<dim3(3072 / 32, 2048 / 32), 256, 0, stream>>>(wqkv, wqkvT, 2048, 3072);
    cvt_transpose_kernel<<<dim3(2048 / 32, 2048 / 32), 256, 0, stream>>>(wout, woutT, 2048, 2048);

    gemm_bf16_tn<<<dim3(3072 / 128, 4096 / 128), 256, 0, stream>>>(
        xbf, wqkvT, qkvf, 4096, 3072, 2048);

    rope_split_kernel<<<4096, 256, 0, stream>>>(qkvf, Qb, Kb, VTb);

    attn_kernel<<<1024, 256, 0, stream>>>(Qb, Kb, VTb, attnO);

    gemm_bf16_tn<<<dim3(2048 / 128, 4096 / 128), 256, 0, stream>>>(
        attnO, woutT, out, 4096, 2048, 2048);
}
